// MHDPAttention_7301444403568
// MI455X (gfx1250) — matrix operand order repaired and checked by reading, not compiled, not run on hardware
//
#include <hip/hip_runtime.h>
#include <hip/hip_bf16.h>
#include <stdint.h>

// ---------------------------------------------------------------------------
// MHDPA forward for MI455X (gfx1250, wave32, WMMA bf16 + TDM).
//   B=8, L=1024, E=256 (head dim), H=12
//   k1: qkv = x @ W_qkv + b   -> qk bf16 [8192,6144], V transposed bf16 [96,256,1024]
//   k2: flash attention, TDM double-buffered K/V staging -> attn bf16 [8192,3072]
//   k3: out = attn @ W_out    -> fp32 [8192,256]
// ---------------------------------------------------------------------------

typedef __attribute__((ext_vector_type(16))) __bf16 v16bf;
typedef __attribute__((ext_vector_type(8)))  float  v8f;
typedef __attribute__((ext_vector_type(4)))  unsigned uv4;
typedef __attribute__((ext_vector_type(8)))  int      iv8;
typedef __attribute__((ext_vector_type(4)))  int      iv4;

union FragU {
  v16bf v;
  uint4 q[2];
  unsigned short h[16];
};

__device__ __forceinline__ unsigned short f2bf(float f) {
  unsigned u = __float_as_uint(f);
  unsigned r = 0x7FFFu + ((u >> 16) & 1u);   // round-to-nearest-even
  return (unsigned short)((u + r) >> 16);
}
__device__ __forceinline__ unsigned pack2(float lo, float hi) {
  return (unsigned)f2bf(lo) | ((unsigned)f2bf(hi) << 16);
}

__device__ __forceinline__ v8f v8f_zero() {
  v8f z;
#pragma unroll
  for (int i = 0; i < 8; ++i) z[i] = 0.f;
  return z;
}

__device__ __forceinline__ v8f wmma_bf16(v16bf a, v16bf b, v8f c) {
  return __builtin_amdgcn_wmma_f32_16x16x32_bf16(
      false, a, false, b, (short)0, c, false, false);
}

// A fragment: 16x32 bf16 tile from row-major LDS, lda elems.
// lane m=lane&15, hf=lane>>4; elems 0..7 = K hf*8..+7, 8..15 = K 16+hf*8..+7.
__device__ __forceinline__ v16bf frag_a_rm(const unsigned short* base, int lda) {
  int lane = threadIdx.x & 31;
  int m = lane & 15, hf = lane >> 4;
  const unsigned short* row = base + m * lda;
  FragU f;
  f.q[0] = *(const uint4*)(row + hf * 8);
  f.q[1] = *(const uint4*)(row + 16 + hf * 8);
  return f.v;
}

// B fragment from TRANSPOSED LDS store Bt[n][k] (k=0..31 at base), ldt elems.
// lane n=lane&15, hf=lane>>4; elems 0..7 = K hf*8..+7, 8..15 = K 16+hf*8..+7 (2x b128).
__device__ __forceinline__ v16bf frag_b_tr(const unsigned short* base, int ldt) {
  int lane = threadIdx.x & 31;
  int n = lane & 15, hf = lane >> 4;
  const unsigned short* row = base + n * ldt;
  FragU f;
  f.q[0] = *(const uint4*)(row + hf * 8);
  f.q[1] = *(const uint4*)(row + 16 + hf * 8);
  return f.v;
}

// ---------------------------------------------------------------------------
// Tensor Data Mover: DMA a (hgt rows x w elems) bf16 tile, row stride `stride`
// elems, from global into LDS (packed rows). D# layout per CDNA5 ISA §8.3/8.4.
// ---------------------------------------------------------------------------
#if defined(__has_builtin)
#if __has_builtin(__builtin_amdgcn_tensor_load_to_lds) && \
    __has_builtin(__builtin_amdgcn_s_wait_tensorcnt)
#define USE_TDM 1
#endif
#endif

#ifndef USE_TDM
#define USE_TDM 0
#endif

#if USE_TDM
__device__ __forceinline__ void tdm_load_2d(unsigned lds_byte_off,
                                            const unsigned short* src,
                                            unsigned w, unsigned hgt,
                                            unsigned stride) {
  unsigned long long ga = (unsigned long long)(size_t)src;
  uv4 g0;
  g0[0] = 1u;                                    // count=1, user mode
  g0[1] = lds_byte_off;                          // lds_addr
  g0[2] = (unsigned)ga;                          // global_addr[31:0]
  g0[3] = (unsigned)((ga >> 32) & 0x01FFFFFFull) // global_addr[56:32]
          | (2u << 30);                          // type=2 ("image")
  iv8 g1;
  g1[0] = (int)(1u << 16);                       // wg_mask=0, data_size=2B
  g1[1] = (int)((w & 0xFFFFu) << 16);            // tensor_dim0[15:0]
  g1[2] = (int)(((w >> 16) & 0xFFFFu) | ((hgt & 0xFFFFu) << 16));
  g1[3] = (int)(((hgt >> 16) & 0xFFFFu) | ((w & 0xFFFFu) << 16)); // tile_dim0=w
  g1[4] = (int)(hgt & 0xFFFFu);                  // tile_dim1=hgt, tile_dim2=0
  g1[5] = (int)stride;                           // tensor_dim0_stride[31:0]
  g1[6] = 0;
  g1[7] = 0;
  iv4 z4 = {0, 0, 0, 0};                         // groups 2/3: unused (2D)
#if __clang_major__ >= 23
  iv8 z8 = {0, 0, 0, 0, 0, 0, 0, 0};
  __builtin_amdgcn_tensor_load_to_lds(g0, g1, z4, z4, z8, 0);
#else
  __builtin_amdgcn_tensor_load_to_lds(g0, g1, z4, z4, 0);
#endif
}
#endif

// ---------------------------------------------------------------------------
// GEMM: C[M,N] = A[M,K] @ B[K,N] (+bias). WMMA bf16, fp32 accum.
// Block tile 64(M) x 128(N) x 32(K), 256 threads = 8 waves (4x2).
// If vtsplit != nullptr, columns n>=6144 are stored TRANSPOSED as
// vtsplit[(b*12+h)*256+e][l] (b=m/1024, l=m%1024, h=(n-6144)/256, e=(n-6144)%256)
// using one b128 store per lane per tile (C-layout rows are consecutive l).
// ---------------------------------------------------------------------------
template <bool A_BF16, bool OUT_BF16>
__global__ __launch_bounds__(256) void gemm_bias_wmma(
    const void* __restrict__ Av, const float* __restrict__ Bm,
    const float* __restrict__ bias, void* __restrict__ Cv,
    int M, int N, int K, int ldc, unsigned short* __restrict__ vtsplit) {
  __shared__ __align__(16) unsigned short As[64 * 32];    // 4 KB
  __shared__ __align__(16) unsigned short Bt[128 * 32];   // 8 KB, transposed

  const int tid = threadIdx.x;
  const int m0 = blockIdx.y * 64;
  const int n0 = blockIdx.x * 128;

  const int wv = tid >> 5, lane = tid & 31;
  const int wm = wv >> 1, wn = wv & 1;
  const int hf = lane >> 4, nl = lane & 15;

  v8f acc[4];
#pragma unroll
  for (int j = 0; j < 4; ++j) acc[j] = v8f_zero();

  const int arow = tid >> 2, ac0 = (tid & 3) * 8;
  const int bkr = tid >> 3, bn0 = (tid & 7) * 16;

  for (int k0 = 0; k0 < K; k0 += 32) {
    __syncthreads();
    if (A_BF16) {
      const unsigned short* A = (const unsigned short*)Av;
      *(uint4*)&As[arow * 32 + ac0] =
          *(const uint4*)(A + (size_t)(m0 + arow) * K + k0 + ac0);
    } else {
      const float* A = (const float*)Av;
      const float* src = A + (size_t)(m0 + arow) * K + k0 + ac0;
      float4 f0 = *(const float4*)src;
      float4 f1 = *(const float4*)(src + 4);
      uint4 u;
      u.x = pack2(f0.x, f0.y); u.y = pack2(f0.z, f0.w);
      u.z = pack2(f1.x, f1.y); u.w = pack2(f1.z, f1.w);
      *(uint4*)&As[arow * 32 + ac0] = u;
    }
    {
      const float* src = Bm + (size_t)(k0 + bkr) * N + n0 + bn0;
      float4 gv4[4];
      gv4[0] = *(const float4*)(src);
      gv4[1] = *(const float4*)(src + 4);
      gv4[2] = *(const float4*)(src + 8);
      gv4[3] = *(const float4*)(src + 12);
      const float* gv = (const float*)gv4;
#pragma unroll
      for (int i = 0; i < 16; ++i)
        Bt[(bn0 + i) * 32 + bkr] = f2bf(gv[i]);
    }
    __syncthreads();

    v16bf a = frag_a_rm(&As[wm * 16 * 32], 32);
#pragma unroll
    for (int j = 0; j < 4; ++j) {
      v16bf b = frag_b_tr(&Bt[(wn * 64 + j * 16) * 32], 32);
      acc[j] = wmma_bf16(a, b, acc[j]);
    }
  }

  // ---- epilogue ----
  if constexpr (OUT_BF16) {
    if (vtsplit != nullptr && n0 >= 6144) {
      // V panel: transposed b128 stores into Vt[b,h,e,l]
      int bb = m0 >> 10;
      int l0 = (m0 & 1023) + wm * 16 + 8 * hf;
#pragma unroll
      for (int j = 0; j < 4; ++j) {
        int ng = n0 + wn * 64 + j * 16 + nl;
        float bv = bias[ng];
        int hh = (ng - 6144) >> 8, e = (ng - 6144) & 255;
        union { unsigned short s[8]; uint4 q; } tmp;
#pragma unroll
        for (int r = 0; r < 8; ++r) tmp.s[r] = f2bf(acc[j][r] + bv);
        *(uint4*)&vtsplit[((size_t)(bb * 12 + hh) * 256 + e) * 1024 + l0] =
            tmp.q;
      }
    } else {
#pragma unroll
      for (int j = 0; j < 4; ++j) {
        int ng = n0 + wn * 64 + j * 16 + nl;
        float bv = (bias != nullptr) ? bias[ng] : 0.f;
#pragma unroll
        for (int r = 0; r < 8; ++r) {
          int mg = m0 + wm * 16 + 8 * hf + r;
          ((unsigned short*)Cv)[(size_t)mg * ldc + ng] = f2bf(acc[j][r] + bv);
        }
      }
    }
  } else {
#pragma unroll
    for (int j = 0; j < 4; ++j) {
      int ng = n0 + wn * 64 + j * 16 + nl;
      float bv = (bias != nullptr) ? bias[ng] : 0.f;
#pragma unroll
      for (int r = 0; r < 8; ++r) {
        int mg = m0 + wm * 16 + 8 * hf + r;
        ((float*)Cv)[(size_t)mg * ldc + ng] = acc[j][r] + bv;
      }
    }
  }
}

// ---------------------------------------------------------------------------
// Fused attention: one workgroup per (b, h, 64-row Q tile).
// Online-softmax over 16 key blocks of 64; K/V TDM-staged with double
// buffering (DMA for block kb+1 overlaps WMMA on block kb). ~185 KB LDS.
// NOTE: all LDS pointers are derived by plain pointer arithmetic from the
// extern-shared base (no pointer arrays!) so InferAddressSpaces keeps them
// addrspace(3) and fragment loads lower to ds_load_b128, not flat_load.
// ---------------------------------------------------------------------------
#define LQK 6144
#define LATT 3072

// LDS layout (elements of unsigned short unless noted):
//   Qs   @ 0          64*256
//   Ks0  @ 16384      64*256
//   Ks1  @ 32768      64*256
//   Vt0  @ 49152      256*64
//   Vt1  @ 65536      256*64
//   Ps   @ 81920      64*64
//   Sf   @ 86016      64*64 floats
//   mrow/lrow/crow after Sf
#define QS_E   0
#define KS_E   (64 * 256)
#define VT_E   (3 * 64 * 256)
#define PS_E   (5 * 64 * 256)
#define SF_E   (5 * 64 * 256 + 64 * 64)

__global__ __launch_bounds__(256) void attention_fwd(
    const unsigned short* __restrict__ qk,
    const unsigned short* __restrict__ vt,
    unsigned short* __restrict__ attn) {
  extern __shared__ __align__(16) unsigned char smem_raw[];
  unsigned short* lds = (unsigned short*)smem_raw;
  unsigned short* Qs = lds + QS_E;
  unsigned short* Ps = lds + PS_E;
  float* Sf = (float*)(lds + SF_E);
  float* mrow = Sf + 64 * 64;
  float* lrow = mrow + 64;
  float* crow = lrow + 64;

  const int tid = threadIdx.x;
  const int b = blockIdx.z, h = blockIdx.y;
  const int q0 = blockIdx.x * 64;

  const unsigned short* qbase = qk + (size_t)b * 1024 * LQK + (size_t)h * 256;
  const unsigned short* kbase = qbase + 3072;
  const unsigned short* vtbase = vt + (size_t)(b * 12 + h) * 256 * 1024;

  const int wv = tid >> 5, lane = tid & 31;
  const int hf = lane >> 4, nl = lane & 15;

#if USE_TDM
  // Issue Q tile + first K/V block; completion awaited inside iteration 0.
  if (wv == 0) {
    tdm_load_2d(QS_E * 2, qbase + (size_t)q0 * LQK, 256, 64, LQK);
    tdm_load_2d(KS_E * 2, kbase, 256, 64, LQK);
    tdm_load_2d(VT_E * 2, vtbase, 64, 256, 1024);
  }
#else
  {
    int row = tid >> 2, seg = (tid & 3) * 64;
    const unsigned short* src = qbase + (size_t)(q0 + row) * LQK + seg;
    unsigned short* dst = Qs + row * 256 + seg;
#pragma unroll
    for (int i = 0; i < 8; ++i)
      *(uint4*)(dst + i * 8) = *(const uint4*)(src + i * 8);
  }
#endif
  if (tid < 64) { mrow[tid] = -1e30f; lrow[tid] = 0.f; }

  v8f accO[8];
#pragma unroll
  for (int j = 0; j < 8; ++j) accO[j] = v8f_zero();

  const int sm = wv & 3;    // S phase: 4x2 wave grid over 64x64 scores
  const int snb = wv >> 2;
  const int wm = wv >> 1;   // O phase: 4x2 wave grid over 64x256 output
  const int wn = wv & 1;

  for (int kb = 0; kb < 16; ++kb) {
    const int cur = kb & 1;
    // LDS pointers via arithmetic from the shared base (stays addrspace(3))
    const unsigned short* Ks = lds + KS_E + cur * (64 * 256);
    const unsigned short* Vl = lds + VT_E + cur * (256 * 64);
    __syncthreads();   // everyone done reading buffers[cur] from kb-2
#if USE_TDM
    if (wv == 0) {
      if (kb < 15) {
        // prefetch next block into the other buffer, then wait only for the
        // current block's two tiles (next two stay in flight during compute)
        const unsigned ksn = (KS_E + (cur ^ 1) * (64 * 256)) * 2;
        const unsigned vtn = (VT_E + (cur ^ 1) * (256 * 64)) * 2;
        tdm_load_2d(ksn, kbase + (size_t)((kb + 1) * 64) * LQK, 256, 64, LQK);
        tdm_load_2d(vtn, vtbase + (kb + 1) * 64, 64, 256, 1024);
        __builtin_amdgcn_s_wait_tensorcnt(2);
      } else {
        __builtin_amdgcn_s_wait_tensorcnt(0);
      }
    }
#else
    {
      int row = tid >> 2, seg = (tid & 3) * 64;
      const unsigned short* ksrc = kbase + (size_t)(kb * 64 + row) * LQK + seg;
      unsigned short* kdst = lds + KS_E + cur * (64 * 256) + row * 256 + seg;
#pragma unroll
      for (int i = 0; i < 8; ++i)
        *(uint4*)(kdst + i * 8) = *(const uint4*)(ksrc + i * 8);
      // V transposed tile: row d = tid, 64 contiguous l values
      const unsigned short* vsrc = vtbase + (size_t)tid * 1024 + kb * 64;
      unsigned short* vdst = lds + VT_E + cur * (256 * 64) + tid * 64;
#pragma unroll
      for (int i = 0; i < 8; ++i)
        *(uint4*)(vdst + i * 8) = *(const uint4*)(vsrc + i * 8);
    }
#endif
    __syncthreads();   // publish buffers[cur]

    // ---- S = (Q K^T) * scale : wave computes a 16x32 strip of 64x64 ----
    v8f accS[2];
    accS[0] = v8f_zero();
    accS[1] = v8f_zero();
#pragma unroll
    for (int dc = 0; dc < 8; ++dc) {
      v16bf a = frag_a_rm(Qs + sm * 16 * 256 + dc * 32, 256);
      v16bf b0 = frag_b_tr(Ks + (snb * 32 + 0) * 256 + dc * 32, 256);
      accS[0] = wmma_bf16(a, b0, accS[0]);
      v16bf b1 = frag_b_tr(Ks + (snb * 32 + 16) * 256 + dc * 32, 256);
      accS[1] = wmma_bf16(a, b1, accS[1]);
    }
#pragma unroll
    for (int t = 0; t < 2; ++t) {
      int n = snb * 32 + t * 16 + nl;
#pragma unroll
      for (int r = 0; r < 8; ++r) {
        int m = sm * 16 + 8 * hf + r;
        Sf[m * 64 + n] = accS[t][r] * 0.0625f;   // 1/sqrt(256)
      }
    }
    __syncthreads();

    // ---- online softmax: one thread per q row ----
    if (tid < 64) {
      int row = tid;
      float om = mrow[row];
      float rmax = -1e30f;
#pragma unroll 8
      for (int j2 = 0; j2 < 64; ++j2) rmax = fmaxf(rmax, Sf[row * 64 + j2]);
      float nm = fmaxf(om, rmax);
      float c = __expf(om - nm);
      float sum = 0.f;
#pragma unroll 8
      for (int j2 = 0; j2 < 64; ++j2) {
        float p = __expf(Sf[row * 64 + j2] - nm);
        sum += p;
        Ps[row * 64 + j2] = f2bf(p);
      }
      lrow[row] = lrow[row] * c + sum;
      mrow[row] = nm;
      crow[row] = c;
    }
    __syncthreads();

    // ---- rescale running output, then accumulate P @ V ----
    float cv[8];
#pragma unroll
    for (int r = 0; r < 8; ++r) cv[r] = crow[wm * 16 + 8 * hf + r];
#pragma unroll
    for (int j = 0; j < 8; ++j)
#pragma unroll
      for (int r = 0; r < 8; ++r) accO[j][r] *= cv[r];

#pragma unroll
    for (int kkc = 0; kkc < 2; ++kkc) {
      v16bf a = frag_a_rm(Ps + wm * 16 * 64 + kkc * 32, 64);
#pragma unroll
      for (int j = 0; j < 8; ++j) {
        // Vt LDS tile is [d][kk] -> exactly Bt layout, 2x b128 per frag
        v16bf bb = frag_b_tr(Vl + (wn * 128 + j * 16) * 64 + kkc * 32, 64);
        accO[j] = wmma_bf16(a, bb, accO[j]);
      }
    }
  }
  __syncthreads();

  // ---- epilogue: divide by row sums, store bf16 [B*L, H*E] ----
  float lv[8];
#pragma unroll
  for (int r = 0; r < 8; ++r) lv[r] = 1.0f / lrow[wm * 16 + 8 * hf + r];
#pragma unroll
  for (int j = 0; j < 8; ++j) {
    int n = wn * 128 + j * 16 + nl;
#pragma unroll
    for (int r = 0; r < 8; ++r) {
      int m = q0 + wm * 16 + 8 * hf + r;
      attn[(size_t)(b * 1024 + m) * LATT + (size_t)h * 256 + n] =
          f2bf(accO[j][r] * lv[r]);
    }
  }
}

// ---------------------------------------------------------------------------
extern "C" void kernel_launch(void* const* d_in, const int* in_sizes, int n_in,
                              void* d_out, int out_size, void* d_ws,
                              size_t ws_size, hipStream_t stream) {
  (void)in_sizes; (void)n_in; (void)out_size; (void)ws_size;
  const float* x    = (const float*)d_in[0];  // [8192, 256]
  const float* Wqkv = (const float*)d_in[1];  // [256, 9216]
  const float* bqkv = (const float*)d_in[2];  // [9216]
  const float* Wout = (const float*)d_in[3];  // [3072, 256]
  float* out = (float*)d_out;                 // [8192, 256]

  unsigned short* qk   = (unsigned short*)d_ws;      // bf16 [8192, 6144]
  unsigned short* vtb  = qk + (size_t)8192 * 6144;   // bf16 [96*256, 1024]
  unsigned short* attn = vtb + (size_t)96 * 256 * 1024;  // bf16 [8192, 3072]

  // 1) qkv = x @ W_qkv + b   (Q/K panels row-major, V panel transposed)
  gemm_bias_wmma<false, true>
      <<<dim3(9216 / 128, 8192 / 64), 256, 0, stream>>>(
          x, Wqkv, bqkv, qk, 8192, 9216, 256, 6144, vtb);

  // 2) fused multi-head attention (TDM double-buffered staging)
  constexpr size_t SMEM =
      (size_t)(64 * 256 * 5 + 64 * 64) * 2 + (size_t)(64 * 64 + 3 * 64) * 4;
  hipFuncSetAttribute((const void*)attention_fwd,
                      hipFuncAttributeMaxDynamicSharedMemorySize, (int)SMEM);
  attention_fwd<<<dim3(16, 12, 8), 256, SMEM, stream>>>(qk, vtb, attn);

  // 3) out = attn @ W_out
  gemm_bias_wmma<true, false>
      <<<dim3(256 / 128, 8192 / 64), 256, 0, stream>>>(
          attn, Wout, nullptr, out, 8192, 256, 3072, 256, nullptr);
}
